// SimpleVoiceCloneModel_25984552141470
// MI455X (gfx1250) — matrix operand order repaired and checked by reading, not compiled, not run on hardware
//
#include <hip/hip_runtime.h>

// ---------------------------------------------------------------------------
// Types / helpers
// ---------------------------------------------------------------------------
typedef __attribute__((ext_vector_type(16))) __bf16         v16bf;
typedef __attribute__((ext_vector_type(8)))  float          v8f;
typedef __attribute__((ext_vector_type(8)))  unsigned short v8us;

union FragBF {
    v16bf bf;
    v8us  us[2];
};

__device__ __forceinline__ unsigned short f2bf(float f) {
    unsigned int u = __float_as_uint(f);
    u += 0x7FFFu + ((u >> 16) & 1u);      // round-to-nearest-even
    return (unsigned short)(u >> 16);
}
__device__ __forceinline__ float bf2f(unsigned short s) {
    return __uint_as_float(((unsigned int)s) << 16);
}
__device__ __forceinline__ float sigmf(float x) {
    return 1.0f / (1.0f + __expf(-x));
}

// ---------------------------------------------------------------------------
// Generic bf16 WMMA GEMM:  out[M,N] = A[M,Kpad] @ W[N,Kpad]^T (+bias0+bias1)
// Each wave computes a 16 x (16*NSUB) tile: A fragment loaded once per
// k-block, reused for NSUB WMMAs.  Requires (N/16) % NSUB == 0.
// If permT > 0, output rows are remapped (b*T+t) -> (t*B+b)  (T = permT),
// producing a [T, B, ldc] tensor that the LSTM streams contiguously per step.
// ---------------------------------------------------------------------------
template <int NSUB>
__global__ void gemm_bf16_wmma(const unsigned short* __restrict__ A,
                               const unsigned short* __restrict__ W,
                               const float* __restrict__ bias0,
                               const float* __restrict__ bias1,
                               float* __restrict__ outF,
                               unsigned short* __restrict__ outB,
                               int M, int N, int Kpad, int ldc, int ncol0,
                               int relu, int permT)
{
    const int lane        = threadIdx.x & 31;
    const int waveInBlk   = threadIdx.x >> 5;
    const int wavesPerBlk = blockDim.x >> 5;
    const long long wid   = (long long)blockIdx.x * wavesPerBlk + waveInBlk;
    const int ngroups     = (N >> 4) / NSUB;
    const long long total = (long long)(M >> 4) * ngroups;
    if (wid >= total) return;                 // wave-uniform
    const int mt  = (int)(wid / ngroups);
    const int nt0 = (int)(wid % ngroups) * NSUB;
    const int lo = lane & 15, hi = lane >> 4;

    v8f acc[NSUB];
#pragma unroll
    for (int g = 0; g < NSUB; ++g) acc[g] = (v8f){};

    const unsigned short* arow  = A + (size_t)(mt * 16 + lo) * Kpad;
    const unsigned short* wrow0 = W + (size_t)(nt0 * 16 + lo) * Kpad;
    for (int kb = 0; kb < Kpad; kb += 32) {
        FragBF a;
        a.us[0] = *(const v8us*)(arow + kb + 8 * hi);
        a.us[1] = *(const v8us*)(arow + kb + 8 * hi + 16);
#pragma unroll
        for (int g = 0; g < NSUB; ++g) {
            FragBF b;
            const unsigned short* wrow = wrow0 + (size_t)g * 16 * Kpad;
            b.us[0] = *(const v8us*)(wrow + kb + 8 * hi);
            b.us[1] = *(const v8us*)(wrow + kb + 8 * hi + 16);
            acc[g] = __builtin_amdgcn_wmma_f32_16x16x32_bf16(false, a.bf, false, b.bf,
                                                             (short)0, acc[g], false, false);
        }
    }

    const int Brows = permT > 0 ? M / permT : 0;
#pragma unroll
    for (int g = 0; g < NSUB; ++g) {
        const int n = (nt0 + g) * 16 + lo;
        float bsum = 0.0f;
        if (bias0) bsum += bias0[n];
        if (bias1) bsum += bias1[n];
#pragma unroll
        for (int r = 0; r < 8; ++r) {
            const int m = mt * 16 + r + 8 * hi;
            float v = acc[g][r] + bsum;
            if (relu) v = fmaxf(v, 0.0f);
            size_t orow = (size_t)m;
            if (permT > 0) {                       // (b*T+t) -> (t*B+b)
                const int b = m / permT;
                const int t = m - b * permT;
                orow = (size_t)t * Brows + b;
            }
            const size_t o = orow * ldc + ncol0 + n;
            if (outF) outF[o] = v;
            else      outB[o] = f2bf(v);
        }
    }
}

// ---------------------------------------------------------------------------
// Whole-sequence fused LSTM layer.
//   grid  = 8 workgroups  (mt: 16 batch rows each)
//   block = 512 threads = 16 waves (jt: 16 hidden columns each)
// Hidden state: double-buffered bf16 tile in LDS (2 x 16 x 256).
// Cell state + context accumulator: registers for the entire sequence.
// gx is [T, B, 1024]: each step reads one contiguous 512 KB slice (prefetched).
// W_hh fragments are re-read from L2 each step; an opaque zero byte-offset
// (asm in-out operand) makes the addresses loop-variant so the loads are not
// hoisted (which previously caused scratch spills), while keeping the base
// pointer's global-address-space provenance (global_load, not flat_load).
// ---------------------------------------------------------------------------
__global__ void __launch_bounds__(512)
lstm_seq_wmma(const float* __restrict__ gx,            // [T,B,1024]
              const unsigned short* __restrict__ Whh,   // [1024,256] bf16
              unsigned short* __restrict__ hs_out,      // [B,T,256] bf16 or null
              float* __restrict__ ctx_out,              // [128,256] or null
              int T)
{
    __shared__ __align__(16) unsigned short hbuf[2][16][256];

    const int lane = threadIdx.x & 31;
    const int jt   = threadIdx.x >> 5;   // wave index in block = n-tile
    const int lo = lane & 15, hi = lane >> 4;
    const int mt = blockIdx.x;           // 0..7

    // zero initial hidden state (buffer 0)
    for (int i = threadIdx.x; i < 16 * 256 / 8; i += blockDim.x)
        ((v8us*)&hbuf[0][0][0])[i] = (v8us){};

    float creg[8];
    float ctxreg[8];
#pragma unroll
    for (int r = 0; r < 8; ++r) { creg[r] = 0.0f; ctxreg[r] = 0.0f; }

    const int j = jt * 16 + lo;
    size_t woff = 0;                     // opaque zero offset (defeats LICM only)
    __syncthreads();

    for (int t = 0; t < T; ++t) {
        asm("" : "+s"(woff));            // loop-variant, provenance preserved
        const unsigned short* Wt = Whh + woff;
        const unsigned short* hprev = &hbuf[t & 1][0][0];
        unsigned short*       hnext = &hbuf[(t & 1) ^ 1][0][0];

        // prefetch next timestep's contiguous gx slice (64KB for this block)
        if (t + 1 < T)
            __builtin_prefetch(gx + ((size_t)(t + 1) * 128 + mt * 16) * 1024
                                  + (size_t)threadIdx.x * 32, 0, 0);

        v8f acc[4];
#pragma unroll
        for (int g = 0; g < 4; ++g) acc[g] = (v8f){};

        const unsigned short* arow = hprev + (size_t)lo * 256;
#pragma unroll
        for (int kb = 0; kb < 256; kb += 32) {
            FragBF a;
            a.us[0] = *(const v8us*)(arow + kb + 8 * hi);
            a.us[1] = *(const v8us*)(arow + kb + 8 * hi + 16);
#pragma unroll
            for (int g = 0; g < 4; ++g) {
                FragBF b;
                const unsigned short* wrow = Wt + (size_t)(g * 256 + jt * 16 + lo) * 256;
                b.us[0] = *(const v8us*)(wrow + kb + 8 * hi);
                b.us[1] = *(const v8us*)(wrow + kb + 8 * hi + 16);
                acc[g] = __builtin_amdgcn_wmma_f32_16x16x32_bf16(false, a.bf, false, b.bf,
                                                                 (short)0, acc[g], false, false);
            }
        }

#pragma unroll
        for (int r = 0; r < 8; ++r) {
            const int ml = r + 8 * hi;                 // row within this m-tile
            const int m  = mt * 16 + ml;               // global batch row
            const size_t gxo = ((size_t)t * 128 + m) * 1024 + j;
            const float iv = acc[0][r] + gx[gxo];
            const float fv = acc[1][r] + gx[gxo + 256];
            const float gv = acc[2][r] + gx[gxo + 512];
            const float ov = acc[3][r] + gx[gxo + 768];
            const float cn = sigmf(fv) * creg[r] + sigmf(iv) * tanhf(gv);
            const float hn = sigmf(ov) * tanhf(cn);
            creg[r] = cn;
            ctxreg[r] += hn;
            const unsigned short hb = f2bf(hn);
            hnext[(size_t)ml * 256 + j] = hb;
            if (hs_out)
                hs_out[((size_t)m * T + t) * 256 + j] = hb;
        }
        __syncthreads();
    }

    if (ctx_out) {
#pragma unroll
        for (int r = 0; r < 8; ++r) {
            const int m = mt * 16 + r + 8 * hi;
            ctx_out[(size_t)m * 256 + j] = ctxreg[r];
        }
    }
}

// ---------------------------------------------------------------------------
// Packing / elementwise kernels
// ---------------------------------------------------------------------------
__global__ void pack_weight(const float* __restrict__ src, unsigned short* __restrict__ dst,
                            int N, int Kin, int Kpad)
{
    const long long n = (long long)N * Kpad;
    long long i = (long long)blockIdx.x * blockDim.x + threadIdx.x;
    if (i >= n) return;
    const int k = (int)(i % Kpad);
    const long long r = i / Kpad;
    dst[i] = f2bf(k < Kin ? src[r * Kin + k] : 0.0f);
}

__global__ void embed_pack(const int* __restrict__ text, const float* __restrict__ emb,
                           unsigned short* __restrict__ out, long long n)
{
    long long i = (long long)blockIdx.x * blockDim.x + threadIdx.x;
    if (i >= n) return;
    const int k = (int)(i & 127);        // E = 128
    const long long row = i >> 7;        // b*512 + t
    const int tok = text[row];
    out[i] = f2bf(emb[(size_t)tok * 128 + k]);
}

__global__ void mel_pack(const float* __restrict__ mels, unsigned short* __restrict__ out,
                         long long n)  // n = 128000*96
{
    long long i = (long long)blockIdx.x * blockDim.x + threadIdx.x;
    if (i >= n) return;
    const int k = (int)(i % 96);
    const long long row = i / 96;        // b*1000 + t
    const int t = (int)(row % 1000);
    float v = 0.0f;
    if (k < 80 && t > 0) v = mels[(row - 1) * 80 + k];
    out[i] = f2bf(v);
}

__global__ void ctx_broadcast(const float* __restrict__ ctx, unsigned short* __restrict__ dec_in,
                              long long n)  // n = 128*1000*256
{
    long long i = (long long)blockIdx.x * blockDim.x + threadIdx.x;
    if (i >= n) return;
    const int j = (int)(i & 255);
    const long long bt = i >> 8;         // b*1000 + t
    const long long b = bt / 1000;
    dec_in[bt * 512 + 256 + j] = f2bf(ctx[b * 256 + j] * (1.0f / 512.0f));
}

__global__ void stop_proj(const unsigned short* __restrict__ h2, const float* __restrict__ sW,
                          const float* __restrict__ sb, float* __restrict__ out, int n)
{
    int i = blockIdx.x * blockDim.x + threadIdx.x;
    if (i >= n) return;
    float s = sb[0];
    const unsigned short* hrow = h2 + (size_t)i * 256;
#pragma unroll 8
    for (int k = 0; k < 256; ++k) s += bf2f(hrow[k]) * sW[k];
    out[i] = sigmf(s);
}

// ---------------------------------------------------------------------------
// Host orchestration
// ---------------------------------------------------------------------------
extern "C" void kernel_launch(void* const* d_in, const int* in_sizes, int n_in,
                              void* d_out, int out_size, void* d_ws, size_t ws_size,
                              hipStream_t stream)
{
    (void)in_sizes; (void)n_in; (void)out_size; (void)ws_size;
    const int Bn = 128, TT = 512, TD = 1000, E = 128, H = 256, G = 1024, MEL = 80, MELP = 96;

    const int*   text  = (const int*)  d_in[0];
    const float* mels  = (const float*)d_in[2];
    const float* emb   = (const float*)d_in[3];
    const float* tWih  = (const float*)d_in[4];
    const float* tWhh  = (const float*)d_in[5];
    const float* tbih  = (const float*)d_in[6];
    const float* tbhh  = (const float*)d_in[7];
    const float* pW1   = (const float*)d_in[8];
    const float* pb1   = (const float*)d_in[9];
    const float* pW2   = (const float*)d_in[10];
    const float* pb2   = (const float*)d_in[11];
    const float* d0Wih = (const float*)d_in[12];
    const float* d0Whh = (const float*)d_in[13];
    const float* d0bih = (const float*)d_in[14];
    const float* d0bhh = (const float*)d_in[15];
    const float* d1Wih = (const float*)d_in[16];
    const float* d1Whh = (const float*)d_in[17];
    const float* d1bih = (const float*)d_in[18];
    const float* d1bhh = (const float*)d_in[19];
    const float* melW  = (const float*)d_in[20];
    const float* melb  = (const float*)d_in[21];
    const float* stW   = (const float*)d_in[22];
    const float* stb   = (const float*)d_in[23];

    float* out_mel  = (float*)d_out;
    float* out_stop = out_mel + (size_t)Bn * TD * MEL;

    // -------- workspace carve-out --------
    char* ws = (char*)d_ws;
    size_t off = 0;
    auto alloc = [&](size_t bytes) -> char* {
        off = (off + 255) & ~(size_t)255;
        char* p = ws + off;
        off += bytes;
        return p;
    };
    float*          gx     = (float*)         alloc((size_t)Bn * TD * G * 4);    // reused 3x
    unsigned short* embB   = (unsigned short*)alloc((size_t)Bn * TT * E * 2);
    unsigned short* decIn  = (unsigned short*)alloc((size_t)Bn * TD * 2 * H * 2);
    unsigned short* pre1   = (unsigned short*)alloc((size_t)Bn * TD * H * 2);
    unsigned short* h1     = (unsigned short*)alloc((size_t)Bn * TD * H * 2);
    unsigned short* h2     = (unsigned short*)alloc((size_t)Bn * TD * H * 2);
    unsigned short* melB   = (unsigned short*)alloc((size_t)Bn * TD * MELP * 2);
    unsigned short* wTih   = (unsigned short*)alloc((size_t)G * E * 2);
    unsigned short* wThh   = (unsigned short*)alloc((size_t)G * H * 2);
    unsigned short* wP1    = (unsigned short*)alloc((size_t)H * MELP * 2);
    unsigned short* wP2    = (unsigned short*)alloc((size_t)H * H * 2);
    unsigned short* wD0ih  = (unsigned short*)alloc((size_t)G * 2 * H * 2);
    unsigned short* wD0hh  = (unsigned short*)alloc((size_t)G * H * 2);
    unsigned short* wD1ih  = (unsigned short*)alloc((size_t)G * H * 2);
    unsigned short* wD1hh  = (unsigned short*)alloc((size_t)G * H * 2);
    unsigned short* wMel   = (unsigned short*)alloc((size_t)MEL * H * 2);
    float*          ctx    = (float*)         alloc((size_t)Bn * H * 4);

    auto cdiv = [](long long a, long long b) { return (unsigned int)((a + b - 1) / b); };

    // -------- weight packing (fp32 -> padded bf16) --------
    auto pk = [&](const float* s, unsigned short* d, int N, int Kin, int Kpad) {
        long long n = (long long)N * Kpad;
        pack_weight<<<cdiv(n, 256), 256, 0, stream>>>(s, d, N, Kin, Kpad);
    };
    pk(tWih,  wTih,  G,   E,     E);
    pk(tWhh,  wThh,  G,   H,     H);
    pk(pW1,   wP1,   H,   MEL,   MELP);
    pk(pW2,   wP2,   H,   H,     H);
    pk(d0Wih, wD0ih, G,   2 * H, 2 * H);
    pk(d0Whh, wD0hh, G,   H,     H);
    pk(d1Wih, wD1ih, G,   H,     H);
    pk(d1Whh, wD1hh, G,   H,     H);
    pk(melW,  wMel,  MEL, H,     H);

    // NSUB-blocked GEMM dispatch: requires (N/16) % NSUB == 0
    auto gemm = [&](const unsigned short* A, const unsigned short* W,
                    const float* b0, const float* b1, float* oF, unsigned short* oB,
                    int M, int N, int Kpad, int ldc, int ncol0, int relu, int permT) {
        const int ntiles = N / 16;
        if (ntiles % 4 == 0) {
            long long waves = (long long)(M / 16) * (ntiles / 4);
            gemm_bf16_wmma<4><<<cdiv(waves, 4), 128, 0, stream>>>(
                A, W, b0, b1, oF, oB, M, N, Kpad, ldc, ncol0, relu, permT);
        } else if (ntiles % 5 == 0) {
            long long waves = (long long)(M / 16) * (ntiles / 5);
            gemm_bf16_wmma<5><<<cdiv(waves, 4), 128, 0, stream>>>(
                A, W, b0, b1, oF, oB, M, N, Kpad, ldc, ncol0, relu, permT);
        } else {
            long long waves = (long long)(M / 16) * ntiles;
            gemm_bf16_wmma<1><<<cdiv(waves, 4), 128, 0, stream>>>(
                A, W, b0, b1, oF, oB, M, N, Kpad, ldc, ncol0, relu, permT);
        }
    };

    // ===================== encoder =====================
    {
        long long n = (long long)Bn * TT * E;
        embed_pack<<<cdiv(n, 256), 256, 0, stream>>>(text, emb, embB, n);
    }
    gemm(embB, wTih, tbih, tbhh, gx, nullptr, Bn * TT, G, E, G, 0, 0, TT);   // -> [TT,B,G]
    lstm_seq_wmma<<<8, 512, 0, stream>>>(gx, wThh, nullptr, ctx, TT);
    {   // context mean -> broadcast into dec_in[:, :, 256:512]
        long long n = (long long)Bn * TD * H;
        ctx_broadcast<<<cdiv(n, 256), 256, 0, stream>>>(ctx, decIn, n);
    }

    // ===================== prenet =====================
    {
        long long n = (long long)Bn * TD * MELP;
        mel_pack<<<cdiv(n, 256), 256, 0, stream>>>(mels, melB, n);
    }
    gemm(melB, wP1, pb1, nullptr, nullptr, pre1,  Bn * TD, H, MELP,  H,     0, 1, 0);
    gemm(pre1, wP2, pb2, nullptr, nullptr, decIn, Bn * TD, H, H,     2 * H, 0, 1, 0);

    // ===================== decoder LSTM 0 =====================
    gemm(decIn, wD0ih, d0bih, d0bhh, gx, nullptr, Bn * TD, G, 2 * H, G, 0, 0, TD); // -> [TD,B,G]
    lstm_seq_wmma<<<8, 512, 0, stream>>>(gx, wD0hh, h1, nullptr, TD);

    // ===================== decoder LSTM 1 =====================
    gemm(h1, wD1ih, d1bih, d1bhh, gx, nullptr, Bn * TD, G, H, G, 0, 0, TD);        // -> [TD,B,G]
    lstm_seq_wmma<<<8, 512, 0, stream>>>(gx, wD1hh, h2, nullptr, TD);

    // ===================== projections =====================
    gemm(h2, wMel, melb, nullptr, out_mel, nullptr, Bn * TD, MEL, H, MEL, 0, 0, 0);
    stop_proj<<<cdiv(Bn * TD, 256), 256, 0, stream>>>(h2, stW, stb, out_stop, Bn * TD);
}
